// SwitchBackLinear_67551245631664
// MI455X (gfx1250) — matrix operand order repaired and checked by reading, compiled, not run on hardware
//
#include <hip/hip_runtime.h>

// ---------------------------------------------------------------------------
// SwitchBackLinear: out = x @ (w_int8 * scale/127)^T + bias
// M = B*S = 4096, K = 4096, N = 4096, fp32 in/out.
//
// int8 weights are exact in bf16; fp32 activations split into hi+lo bf16 and
// both terms accumulated with v_wmma_f32_16x16x32_bf16 (fp32 accumulator)
// -> fp32-accurate result on the CDNA5 matrix pipes.
//
// Fast path (needs 96MB workspace): one-time bf16 prepack, then a GEMM whose
// LDS staging is pure GLOBAL_LOAD_ASYNC_TO_LDS_B128 (ASYNCcnt, GVS form)
// traffic; the compute stage batches all 20 ds_load_b128 fragment loads
// before issuing 16 back-to-back WMMAs.
// Fallback path: fused on-the-fly conversion (no workspace required).
// ---------------------------------------------------------------------------

typedef __bf16 bf16;
typedef __attribute__((ext_vector_type(16))) __bf16 v16bf;
typedef __attribute__((ext_vector_type(8)))  __bf16 v8bf;
typedef __attribute__((ext_vector_type(4)))  __bf16 v4bf;
typedef __attribute__((ext_vector_type(8)))  float  v8f;
typedef __attribute__((ext_vector_type(4)))  float  v4f;
typedef __attribute__((ext_vector_type(4)))  int    v4i;

#define M_DIM 4096
#define N_DIM 4096
#define K_DIM 4096

#define BM 128
#define BN 128
#define BK 32
#define PAD 8
#define LDA (BK + PAD)   // 40 bf16 = 80 B row stride: conflict-free b128 frag loads
#define TB 256

// ---------------------------------------------------------------------------
__device__ __forceinline__ v16bf cat8(v8bf a, v8bf b) {
  return __builtin_shufflevector(a, b, 0, 1, 2, 3, 4, 5, 6, 7,
                                       8, 9, 10, 11, 12, 13, 14, 15);
}

__device__ __forceinline__ void wait_async0() {
#if __has_builtin(__builtin_amdgcn_s_wait_asynccnt)
  __builtin_amdgcn_s_wait_asynccnt(0);
#else
  asm volatile("s_wait_asynccnt 0x0" ::: "memory");
#endif
}

// Async 16B copy global -> LDS, GVS form: SGPR 64-bit base + 32-bit VGPR
// offset. LDS byte offset = low 32 bits of the generic pointer (ISA: LDS
// aperture maps addr[31:0] directly to the wave's LDS offset).
__device__ __forceinline__ void async_copy_b128(const void* sbase, unsigned voff,
                                                void* lptr) {
  unsigned lds = (unsigned)(size_t)lptr;
  asm volatile("global_load_async_to_lds_b128 %0, %1, %2"
               :: "v"(lds), "v"(voff), "s"(sbase) : "memory");
}

// ---- one WMMA K-step (BK=32) over an LDS stage ----------------------------
// Loads ALL fragments first (distinct registers) so the 16 WMMAs issue as a
// contiguous block instead of stalling on dscnt per M-subtile.
__device__ __forceinline__ void wmma_stage(const bf16* sAhiB, const bf16* sAloB,
                                           const bf16* sBB, int waveM, int waveN,
                                           int l16, int lh, v8f (&acc)[4][2]) {
  v16bf bfrag[2], ah[4], al[4];
#pragma unroll
  for (int j = 0; j < 2; ++j) {
    const bf16* p = sBB + (waveN * 32 + j * 16 + l16) * LDA + lh * 8;
    bfrag[j] = cat8(*(const v8bf*)p, *(const v8bf*)(p + 16));
  }
#pragma unroll
  for (int i = 0; i < 4; ++i) {
    const bf16* ph = sAhiB + (waveM * 64 + i * 16 + l16) * LDA + lh * 8;
    const bf16* pl = sAloB + (waveM * 64 + i * 16 + l16) * LDA + lh * 8;
    ah[i] = cat8(*(const v8bf*)ph, *(const v8bf*)(ph + 16));
    al[i] = cat8(*(const v8bf*)pl, *(const v8bf*)(pl + 16));
  }
#pragma unroll
  for (int i = 0; i < 4; ++i) {
#pragma unroll
    for (int j = 0; j < 2; ++j) {
      acc[i][j] = __builtin_amdgcn_wmma_f32_16x16x32_bf16(
          false, ah[i], false, bfrag[j], (short)0, acc[i][j], false, false);
      acc[i][j] = __builtin_amdgcn_wmma_f32_16x16x32_bf16(
          false, al[i], false, bfrag[j], (short)0, acc[i][j], false, false);
    }
  }
}

// ---- epilogue: scale/127 + bias, fp32 out ---------------------------------
__device__ __forceinline__ void store_out(float* __restrict__ out,
                                          const float* __restrict__ scales,
                                          const float* __restrict__ bias,
                                          int mBase, int nBase, int waveM, int waveN,
                                          int l16, int lh, const v8f (&acc)[4][2]) {
  const float inv127 = 1.0f / 127.0f;
#pragma unroll
  for (int j = 0; j < 2; ++j) {
    const int col = nBase + waveN * 32 + j * 16 + l16;
    const float sc = scales[col] * inv127;
    const float bs = bias[col];
#pragma unroll
    for (int i = 0; i < 4; ++i) {
      const int rowBase = mBase + waveM * 64 + i * 16 + lh * 8;
#pragma unroll
      for (int r = 0; r < 8; ++r) {
        out[(size_t)(rowBase + r) * N_DIM + col] = acc[i][j][r] * sc + bs;
      }
    }
  }
}

// ===========================================================================
// Prepack kernels (fast path)
// ===========================================================================
__global__ __launch_bounds__(256)
void prep_x(const float* __restrict__ x, bf16* __restrict__ xhi,
            bf16* __restrict__ xlo) {
  const size_t i = ((size_t)blockIdx.x * 256 + threadIdx.x) * 4;
  v4f v = *(const v4f*)(x + i);
  v4bf h, l;
#pragma unroll
  for (int e = 0; e < 4; ++e) {
    float f = v[e];
    bf16 fh = (bf16)f;
    h[e] = fh;
    l[e] = (bf16)(f - (float)fh);
  }
  *(v4bf*)(xhi + i) = h;
  *(v4bf*)(xlo + i) = l;
}

__global__ __launch_bounds__(256)
void prep_w(const int* __restrict__ wq, bf16* __restrict__ wb) {
  const size_t i = ((size_t)blockIdx.x * 256 + threadIdx.x) * 4;
  v4i v = *(const v4i*)(wq + i);
  v4bf b;
#pragma unroll
  for (int e = 0; e < 4; ++e) b[e] = (bf16)(float)v[e];  // exact: 0..127
  *(v4bf*)(wb + i) = b;
}

// ===========================================================================
// Fast GEMM: prepacked bf16 inputs, async-to-LDS staging
// ===========================================================================
__global__ __launch_bounds__(TB)
void switchback_gemm_async(const bf16* __restrict__ xhi,
                           const bf16* __restrict__ xlo,
                           const bf16* __restrict__ wb,
                           const float* __restrict__ scales,
                           const float* __restrict__ bias,
                           float* __restrict__ out) {
  __shared__ __align__(16) bf16 sAhi[2][BM * LDA];
  __shared__ __align__(16) bf16 sAlo[2][BM * LDA];
  __shared__ __align__(16) bf16 sB  [2][BN * LDA];

  const int tid   = threadIdx.x;
  const int mBase = blockIdx.y * BM;
  const int nBase = blockIdx.x * BN;

  const int lane  = tid & 31;
  const int wave  = tid >> 5;
  const int waveM = wave & 1;
  const int waveN = wave >> 1;
  const int l16   = lane & 15;
  const int lh    = lane >> 4;

  v8f acc[4][2] = {};

  // Per stage: 3 tiles x 8KB = 512 chunks of 16B each; 2 chunks/thread/tile.
  // All offsets fit in 32 bits (arrays are 32 MB) -> GVS addressing.
  auto issueStage = [&](int kt, int buf) {
    const int kBase = kt * BK;
#pragma unroll
    for (int i = 0; i < 2; ++i) {
      int c   = i * TB + tid;
      int row = c >> 2;             // 4 chunks per 32-elem row
      int col = (c & 3) << 3;       // 0,8,16,24 (bf16 elements)
      const unsigned offA = (unsigned)(((mBase + row) * K_DIM + kBase + col) * 2);
      const unsigned offB = (unsigned)(((nBase + row) * K_DIM + kBase + col) * 2);
      const int ls = row * LDA + col;
      async_copy_b128(xhi, offA, &sAhi[buf][ls]);
      async_copy_b128(xlo, offA, &sAlo[buf][ls]);
      async_copy_b128(wb,  offB, &sB  [buf][ls]);
    }
  };

  issueStage(0, 0);
  wait_async0();
  __syncthreads();

  const int KT = K_DIM / BK;   // 128
  for (int kt = 0; kt < KT; ++kt) {
    const int buf = kt & 1;
    if (kt + 1 < KT) issueStage(kt + 1, buf ^ 1);  // DMA overlaps WMMA below
    wmma_stage(&sAhi[buf][0], &sAlo[buf][0], &sB[buf][0],
               waveM, waveN, l16, lh, acc);
    wait_async0();            // this wave's next-stage copies have landed
    __syncthreads();          // everyone's copies have landed
  }

  store_out(out, scales, bias, mBase, nBase, waveM, waveN, l16, lh, acc);
}

// ===========================================================================
// Fallback GEMM: fused on-the-fly conversion (no workspace required)
// ===========================================================================
__global__ __launch_bounds__(TB)
void switchback_gemm_fused(const float* __restrict__ x,
                           const int*   __restrict__ wq,
                           const float* __restrict__ scales,
                           const float* __restrict__ bias,
                           float*       __restrict__ out) {
  __shared__ __align__(16) bf16 sAhi[2][BM * LDA];
  __shared__ __align__(16) bf16 sAlo[2][BM * LDA];
  __shared__ __align__(16) bf16 sB  [2][BN * LDA];

  const int tid   = threadIdx.x;
  const int mBase = blockIdx.y * BM;
  const int nBase = blockIdx.x * BN;

  const int lane  = tid & 31;
  const int wave  = tid >> 5;
  const int waveM = wave & 1;
  const int waveN = wave >> 1;
  const int l16   = lane & 15;
  const int lh    = lane >> 4;

  v8f acc[4][2] = {};
  v4f ax[4];
  v4i bw[4];

  auto loadStage = [&](int kt) {
    const int kBase = kt * BK;
#pragma unroll
    for (int i = 0; i < 4; ++i) {
      int c   = i * TB + tid;
      int row = c >> 3;
      int col = (c & 7) << 2;
      ax[i] = *(const v4f*)(x  + (size_t)(mBase + row) * K_DIM + kBase + col);
      bw[i] = *(const v4i*)(wq + (size_t)(nBase + row) * K_DIM + kBase + col);
    }
  };

  auto storeStage = [&](int buf) {
#pragma unroll
    for (int i = 0; i < 4; ++i) {
      int c   = i * TB + tid;
      int row = c >> 3;
      int col = (c & 7) << 2;
      v4bf h, l, b;
#pragma unroll
      for (int e = 0; e < 4; ++e) {
        float f  = ax[i][e];
        bf16  fh = (bf16)f;
        h[e] = fh;
        l[e] = (bf16)(f - (float)fh);
        b[e] = (bf16)(float)bw[i][e];
      }
      *(v4bf*)&sAhi[buf][row * LDA + col] = h;
      *(v4bf*)&sAlo[buf][row * LDA + col] = l;
      *(v4bf*)&sB  [buf][row * LDA + col] = b;
    }
  };

  loadStage(0);
  storeStage(0);
  __syncthreads();

  const int KT = K_DIM / BK;
  for (int kt = 0; kt < KT; ++kt) {
    const int buf = kt & 1;
    if (kt + 1 < KT) loadStage(kt + 1);
    wmma_stage(&sAhi[buf][0], &sAlo[buf][0], &sB[buf][0],
               waveM, waveN, l16, lh, acc);
    if (kt + 1 < KT) storeStage(buf ^ 1);
    __syncthreads();
  }

  store_out(out, scales, bias, mBase, nBase, waveM, waveN, l16, lh, acc);
}

// ---------------------------------------------------------------------------
extern "C" void kernel_launch(void* const* d_in, const int* in_sizes, int n_in,
                              void* d_out, int out_size, void* d_ws, size_t ws_size,
                              hipStream_t stream) {
  (void)in_sizes; (void)n_in; (void)out_size;
  const float* x      = (const float*)d_in[0];  // (2,2048,4096) fp32
  const int*   wq     = (const int*)  d_in[1];  // (4096,4096) int32
  const float* scales = (const float*)d_in[2];  // (4096,)
  // d_in[3] = weight_fp (unused by reference)
  const float* bias   = (const float*)d_in[4];  // (4096,)
  float* out = (float*)d_out;

  dim3 grid(N_DIM / BN, M_DIM / BM);  // 32 x 32

  const size_t xElems = (size_t)M_DIM * K_DIM;
  const size_t wElems = (size_t)N_DIM * K_DIM;
  const size_t need   = (2 * xElems + wElems) * sizeof(bf16);  // 96 MB

  if (ws_size >= need) {
    bf16* xhi = (bf16*)d_ws;
    bf16* xlo = xhi + xElems;
    bf16* wb  = xlo + xElems;
    prep_x<<<(int)(xElems / 1024), 256, 0, stream>>>(x, xhi, xlo);
    prep_w<<<(int)(wElems / 1024), 256, 0, stream>>>(wq, wb);
    switchback_gemm_async<<<grid, TB, 0, stream>>>(xhi, xlo, wb, scales, bias, out);
  } else {
    switchback_gemm_fused<<<grid, TB, 0, stream>>>(x, wq, scales, bias, out);
  }
}
